// RNN_87849261072899
// MI455X (gfx1250) — matrix operand order repaired and checked by reading, not compiled, not run on hardware
//
#include <hip/hip_runtime.h>

// ============================================================================
// 2-layer GRU x2 streams + MLP head for MI455X (gfx1250, wave32, WMMA).
//
//  1) proj_kernel : xp = x @ Wih^T + bih   (parallel GEMM, WMMA f16->f32,
//                   vectorized staging + packed f32->f16 converts)
//  2) gru_scan    : persistent scan; Whh resident in LDS (f16), 96 WMMA per
//                   timestep, xp streamed via global_load_async_to_lds_b128
//                   double-buffer (ASYNCcnt), fused gate math.
//  3) proj_kernel : layer-1 projection from layer-0 outputs (f16 in).
//  4) gru_scan    : layer 1, emits last hidden state.
//  5) mlp_kernel  : head -> d_out [256] f32.
//
// Workspace (~269 MB): XP_ru | XP_en (f16, 131072x384 each) | Y_ru | Y_en
// (f16, 131072x128 each) | hlast_ru | hlast_en (f32, 256x128 each).
// ============================================================================

typedef __attribute__((ext_vector_type(16))) _Float16 v16h;
typedef __attribute__((ext_vector_type(8)))  _Float16 v8h;
typedef __attribute__((ext_vector_type(8)))  float    v8f;
typedef __attribute__((ext_vector_type(4)))  float    v4f;
typedef __attribute__((ext_vector_type(2)))  float    v2f;

#define GRU_B 256
#define GRU_T 512
#define GRU_I 130
#define GRU_H 128
#define H3    384
#define M_ROWS (GRU_B * GRU_T)   // 131072

union FragU { v16h v; v8h h[2]; };

// A-fragment (16x32 f16): lane<16 -> row M=lane, K {ks*32+0..7, +16..23};
// lane>=16 -> row M=lane-16, K {+8..15, +24..31}.  (ISA 7.12.2)
__device__ __forceinline__ v16h load_a_frag(const _Float16* tile, int stride, int ks) {
    const int lane = threadIdx.x & 31;
    const _Float16* p = tile + (lane & 15) * stride + ks * 32 + ((lane >> 4) << 3);
    FragU u;
    u.h[0] = *(const v8h*)(p);
    u.h[1] = *(const v8h*)(p + 16);
    return u.v;
}

// B-fragment (32x16 f16): lane -> column N=lane&15, K {ks*32+8*(lane>>4)+0..7,
// +16..23} (same K order as the A-fragment).
__device__ __forceinline__ v16h load_b_frag(const _Float16* tile, int stride, int ks) {
    const int lane = threadIdx.x & 31;
    const _Float16* p = tile + (lane & 15) * stride + ks * 32 + ((lane >> 4) << 3);
    FragU u;
    u.h[0] = *(const v8h*)(p);
    u.h[1] = *(const v8h*)(p + 16);
    return u.v;
}

__device__ __forceinline__ v8f wmma_f16(v16h a, v16h b, v8f c) {
    return __builtin_amdgcn_wmma_f32_16x16x32_f16(false, a, false, b, (short)0, c,
                                                  false, false);
}

// packed f32x2 -> f16x2 convert + single 32-bit store (v_cvt_pk_rtz_f16_f32)
__device__ __forceinline__ void store_pk(_Float16* dst, float x, float y) {
    auto pk = __builtin_amdgcn_cvt_pkrtz(x, y);
    union { decltype(pk) p; unsigned u; } cv;
    cv.p = pk;
    *(unsigned*)dst = cv.u;
}

__device__ __forceinline__ float sigmoidf_(float x) {
    return 1.0f / (1.0f + __expf(-x));
}
// tanh via one v_exp_f32 (clamped so exp never overflows)
__device__ __forceinline__ float tanhf_(float x) {
    x = fminf(fmaxf(x, -20.0f), 20.0f);
    float e = __expf(2.0f * x);
    return (e - 1.0f) / (e + 1.0f);
}

// CDNA5 async global->LDS copy (ASYNCcnt-tracked). LDS offset is the low
// 32 bits of the generic pointer (LDS aperture: addr[31:0] == LDS offset).
__device__ __forceinline__ void async_load_b128(unsigned lds_off, const _Float16* gaddr) {
    asm volatile("global_load_async_to_lds_b128 %0, %1, off"
                 :: "v"(lds_off), "v"(gaddr) : "memory");
}

// ----------------------------------------------------------------------------
// proj_kernel: Y[M,384](f16) = f16( X[M,K] @ W[384,K]^T + bias )
// Block 256 thr (8 waves) -> 32 rows x 64 cols, one 16x16 WMMA tile per wave.
// Grid: (384/64, M/32).
// ----------------------------------------------------------------------------
template<typename TIN, int KP>
__global__ __launch_bounds__(256) void proj_kernel(
    const TIN* __restrict__ X, int ldx, int K,
    const float* __restrict__ W, const float* __restrict__ bias,
    _Float16* __restrict__ Y)
{
    constexpr int KPP = KP + 8;     // padded LDS stride (bank-conflict dodge)
    __shared__ __attribute__((aligned(16))) _Float16 sA[32 * KPP];
    __shared__ __attribute__((aligned(16))) _Float16 sB[64 * KPP];

    const int tid  = threadIdx.x;
    const int row0 = blockIdx.y * 32;
    const int col0 = blockIdx.x * 64;

    // ---- stage A tile (f32: 8B vector loads + packed converts; f16: copy)
    if (sizeof(TIN) == 4) {
        const float* Xf = (const float*)X;
        const int K2 = K >> 1;
        for (int idx = tid; idx < 32 * K2; idx += 256) {
            int r = idx / K2, c = idx - r * K2;
            v2f v = *(const v2f*)(Xf + (size_t)(row0 + r) * ldx + 2 * c);
            store_pk(sA + r * KPP + 2 * c, v.x, v.y);
        }
    } else {
        const _Float16* Xh = (const _Float16*)X;
        const int K8 = K >> 3;
        for (int idx = tid; idx < 32 * K8; idx += 256) {
            int r = idx / K8, c = idx - r * K8;
            *(v8h*)(sA + r * KPP + 8 * c) =
                *(const v8h*)(Xh + (size_t)(row0 + r) * ldx + 8 * c);
        }
    }
    if (KP > K) {       // zero pad columns K..KP-1
        const int PW = KP - K;
        for (int idx = tid; idx < 32 * PW; idx += 256) {
            int r = idx / PW, c = K + (idx - r * PW);
            sA[r * KPP + c] = (_Float16)0.0f;
        }
    }
    // ---- stage W tile (f32, rows 8B-aligned)
    {
        const int K2 = K >> 1;
        for (int idx = tid; idx < 64 * K2; idx += 256) {
            int n = idx / K2, c = idx - n * K2;
            v2f v = *(const v2f*)(W + (size_t)(col0 + n) * K + 2 * c);
            store_pk(sB + n * KPP + 2 * c, v.x, v.y);
        }
        if (KP > K) {
            const int PW = KP - K;
            for (int idx = tid; idx < 64 * PW; idx += 256) {
                int n = idx / PW, c = K + (idx - n * PW);
                sB[n * KPP + c] = (_Float16)0.0f;
            }
        }
    }
    __syncthreads();

    const int wave = tid >> 5;
    const int rt = wave >> 2, ct = wave & 3;   // 2 row-tiles x 4 col-tiles
    v8f acc = {};
    #pragma unroll
    for (int ks = 0; ks < KP / 32; ++ks) {
        v16h a = load_a_frag(sA + rt * 16 * KPP, KPP, ks);
        v16h b = load_b_frag(sB + ct * 16 * KPP, KPP, ks);
        acc = wmma_f16(a, b, acc);
    }

    const int lane = tid & 31;
    const int n = col0 + ct * 16 + (lane & 15);
    const float bn = bias[n];
    #pragma unroll
    for (int r = 0; r < 8; ++r) {
        int m = row0 + rt * 16 + ((lane >> 4) << 3) + r;   // C/D lane layout
        Y[(size_t)m * H3 + n] = (_Float16)(acc[r] + bn);
    }
}

// ----------------------------------------------------------------------------
// gru_scan_kernel: 512-step scan. Block owns 16 batch rows of one stream.
// Whh f16 in LDS for the whole kernel; xp double-buffered in LDS via async
// global->LDS loads. Grid: 32 blocks (0-15 stream A, 16-31 stream B).
// ----------------------------------------------------------------------------
#define WPAD 136   // h / Whh LDS stride in halves (128+8 -> distinct banks)
#define GHP  396   // gh staging stride in floats
#define SXPS 384   // xp slice stride in halves (row = 384 = one timestep)

__global__ __launch_bounds__(256) void gru_scan_kernel(
    const _Float16* __restrict__ XPa, const _Float16* __restrict__ XPb,
    const float* __restrict__ Whh_a, const float* __restrict__ Whh_b,
    const float* __restrict__ bhh_a, const float* __restrict__ bhh_b,
    _Float16* __restrict__ Ya, _Float16* __restrict__ Yb,
    float* __restrict__ hla, float* __restrict__ hlb)
{
    __shared__ __attribute__((aligned(16))) _Float16 sW[H3 * WPAD];  // 104448 B
    __shared__ __attribute__((aligned(16))) _Float16 sh[16 * WPAD];  //   4352 B
    __shared__ __attribute__((aligned(16))) float    sgh[16 * GHP];  //  25344 B
    __shared__ __attribute__((aligned(16))) float    sbhh[H3];       //   1536 B
    __shared__ __attribute__((aligned(16))) _Float16 sxp[2 * 16 * SXPS]; // 24576 B

    const int tid = threadIdx.x;
    const int sb  = (int)(blockIdx.x >> 4);          // stream select
    const int b0  = (int)(blockIdx.x & 15) * 16;     // batch-row base

    const _Float16* XP  = sb ? XPb   : XPa;
    const float*    Whh = sb ? Whh_b : Whh_a;
    const float*    bhh = sb ? bhh_b : bhh_a;
    _Float16*       Yo  = sb ? Yb    : Ya;
    float*          hl  = sb ? hlb   : hla;

    // One-time fills: Whh -> LDS f16 (packed converts), h0 = 0, bhh -> LDS.
    for (int idx = tid; idx < H3 * (GRU_H / 2); idx += 256) {
        int n = idx >> 6, c = (idx & 63) * 2;
        v2f v = *(const v2f*)(Whh + (size_t)n * GRU_H + c);
        store_pk(sW + n * WPAD + c, v.x, v.y);
    }
    {
        v8h z = {};
        const int idx = tid;                 // 256 chunks of 8 halves
        *(v8h*)(sh + (idx >> 4) * WPAD + (idx & 15) * 8) = z;
    }
    for (int idx = tid; idx < H3 / 4; idx += 256)
        *(v4f*)(sbhh + idx * 4) = *(const v4f*)(bhh + idx * 4);

    // Async xp feed: 768 16B chunks per timestep slice; 3 chunks per thread.
    unsigned        loff[3];
    const _Float16* gptr[3];
    #pragma unroll
    for (int k = 0; k < 3; ++k) {
        int c  = tid + k * 256;              // 0..767
        int rw = c / 48;                     // batch row 0..15
        int cl = (c - rw * 48) * 8;          // column 0..376
        gptr[k] = XP + (size_t)(b0 + rw) * GRU_T * H3 + cl;
        loff[k] = (unsigned)(uintptr_t)(sxp + rw * SXPS + cl);
    }
    const unsigned bufbytes = 16 * SXPS * 2;
    #pragma unroll
    for (int k = 0; k < 3; ++k) async_load_b128(loff[k], gptr[k]);  // t=0 -> buf 0
    __syncthreads();

    const int wave = tid >> 5;
    const int lane = tid & 31;
    const int gm = tid >> 4;             // gate phase: row 0..15
    const int gj = (tid & 15) << 3;      // gate phase: col base (8 cols/thr)
    _Float16* yrow = Yo ? (Yo + (size_t)(b0 + gm) * GRU_T * GRU_H + gj) : nullptr;

    for (int t = 0; t < GRU_T; ++t) {
        // kick off async loads of the *next* timestep's xp slice
        const int tn = (t + 1 < GRU_T) ? (t + 1) : 0;   // wrap keeps count uniform
        const unsigned dstbuf = (unsigned)((t + 1) & 1) * bufbytes;
        #pragma unroll
        for (int k = 0; k < 3; ++k)
            async_load_b128(loff[k] + dstbuf, gptr[k] + (size_t)tn * H3);

        // ---- gh = h @ Whh^T : each wave owns N-tiles wave*3 + {0,1,2} ----
        v8f acc0 = {}, acc1 = {}, acc2 = {};
        #pragma unroll
        for (int ks = 0; ks < 4; ++ks) {
            v16h a  = load_a_frag(sh, WPAD, ks);
            v16h q0 = load_b_frag(sW + (wave * 3 + 0) * 16 * WPAD, WPAD, ks);
            acc0 = wmma_f16(a, q0, acc0);
            v16h q1 = load_b_frag(sW + (wave * 3 + 1) * 16 * WPAD, WPAD, ks);
            acc1 = wmma_f16(a, q1, acc1);
            v16h q2 = load_b_frag(sW + (wave * 3 + 2) * 16 * WPAD, WPAD, ks);
            acc2 = wmma_f16(a, q2, acc2);
        }
        {   // spill gh to LDS in C/D lane layout
            const int mrow = ((lane >> 4) << 3);
            const int ncol = lane & 15;
            #pragma unroll
            for (int r = 0; r < 8; ++r) {
                sgh[(mrow + r) * GHP + (wave * 3 + 0) * 16 + ncol] = acc0[r];
                sgh[(mrow + r) * GHP + (wave * 3 + 1) * 16 + ncol] = acc1[r];
                sgh[(mrow + r) * GHP + (wave * 3 + 2) * 16 + ncol] = acc2[r];
            }
        }
        // Of the (up to) 6 outstanding async loads, the 3 oldest are this
        // timestep's xp (issued last iteration); async loads complete in
        // order, so <=3 outstanding guarantees they have landed in LDS.
        asm volatile("s_wait_asynccnt 0x3" ::: "memory");
        __syncthreads();

        // ---- fused gates: thread owns (row gm, cols gj..gj+7) ----
        const _Float16* xp = sxp + (t & 1) * (16 * SXPS) + gm * SXPS;
        v8h xr = *(const v8h*)(xp + gj);
        v8h xz = *(const v8h*)(xp + 128 + gj);
        v8h xn = *(const v8h*)(xp + 256 + gj);
        v8h hold = *(const v8h*)(sh + gm * WPAD + gj);

        const float* gb = sgh + gm * GHP;
        v4f gr[2] = { *(const v4f*)(gb + gj),       *(const v4f*)(gb + gj + 4) };
        v4f gz[2] = { *(const v4f*)(gb + 128 + gj), *(const v4f*)(gb + 132 + gj) };
        v4f gn[2] = { *(const v4f*)(gb + 256 + gj), *(const v4f*)(gb + 260 + gj) };
        v4f br[2] = { *(const v4f*)(sbhh + gj),       *(const v4f*)(sbhh + gj + 4) };
        v4f bz[2] = { *(const v4f*)(sbhh + 128 + gj), *(const v4f*)(sbhh + 132 + gj) };
        v4f bn2[2] = { *(const v4f*)(sbhh + 256 + gj), *(const v4f*)(sbhh + 260 + gj) };

        v8h hnew;
        #pragma unroll
        for (int jj = 0; jj < 8; ++jj) {
            const int q = jj >> 2, e = jj & 3;
            float r  = sigmoidf_((float)xr[jj] + gr[q][e] + br[q][e]);
            float z  = sigmoidf_((float)xz[jj] + gz[q][e] + bz[q][e]);
            float nn = tanhf_((float)xn[jj] + r * (gn[q][e] + bn2[q][e]));
            float hv = z * ((float)hold[jj] - nn) + nn;     // (1-z)n + zh
            hnew[jj] = (_Float16)hv;
        }
        *(v8h*)(sh + gm * WPAD + gj) = hnew;
        if (yrow) *(v8h*)(yrow + (size_t)t * GRU_H) = hnew;
        if (hl && t == GRU_T - 1) {
            #pragma unroll
            for (int jj = 0; jj < 8; ++jj)
                hl[(b0 + gm) * GRU_H + gj + jj] = (float)hnew[jj];
        }
        __syncthreads();
    }
}

// ----------------------------------------------------------------------------
// mlp_kernel: out[b] = sigmoid( W2 @ (W1 @ [hru|hen] + b1) + b2 )
// ----------------------------------------------------------------------------
__global__ __launch_bounds__(256) void mlp_kernel(
    const float* __restrict__ hru, const float* __restrict__ hen,
    const float* __restrict__ W1, const float* __restrict__ b1,
    const float* __restrict__ W2, const float* __restrict__ b2,
    float* __restrict__ out)
{
    const int b = threadIdx.x;
    float acc = b2[0];
    for (int k = 0; k < 50; ++k) {
        float s = b1[k];
        const float* w = W1 + k * 256;
        #pragma unroll 8
        for (int j = 0; j < 128; ++j) s += w[j] * hru[b * 128 + j];
        #pragma unroll 8
        for (int j = 0; j < 128; ++j) s += w[128 + j] * hen[b * 128 + j];
        acc += W2[k] * s;
    }
    out[b] = sigmoidf_(acc);
}

// ----------------------------------------------------------------------------
extern "C" void kernel_launch(void* const* d_in, const int* in_sizes, int n_in,
                              void* d_out, int out_size, void* d_ws, size_t ws_size,
                              hipStream_t stream) {
    (void)in_sizes; (void)n_in; (void)out_size; (void)ws_size;

    const float* ru      = (const float*)d_in[0];
    const float* en      = (const float*)d_in[1];
    const float* ru_Wih0 = (const float*)d_in[2];
    const float* ru_Whh0 = (const float*)d_in[3];
    const float* ru_bih0 = (const float*)d_in[4];
    const float* ru_bhh0 = (const float*)d_in[5];
    const float* ru_Wih1 = (const float*)d_in[6];
    const float* ru_Whh1 = (const float*)d_in[7];
    const float* ru_bih1 = (const float*)d_in[8];
    const float* ru_bhh1 = (const float*)d_in[9];
    const float* en_Wih0 = (const float*)d_in[10];
    const float* en_Whh0 = (const float*)d_in[11];
    const float* en_bih0 = (const float*)d_in[12];
    const float* en_bhh0 = (const float*)d_in[13];
    const float* en_Wih1 = (const float*)d_in[14];
    const float* en_Whh1 = (const float*)d_in[15];
    const float* en_bih1 = (const float*)d_in[16];
    const float* en_bhh1 = (const float*)d_in[17];
    const float* W1      = (const float*)d_in[18];
    const float* b1      = (const float*)d_in[19];
    const float* W2      = (const float*)d_in[20];
    const float* b2      = (const float*)d_in[21];
    float* out = (float*)d_out;

    // workspace carve-up
    _Float16* XPr = (_Float16*)d_ws;
    _Float16* XPe = XPr + (size_t)M_ROWS * H3;
    _Float16* Yr  = XPe + (size_t)M_ROWS * H3;
    _Float16* Ye  = Yr  + (size_t)M_ROWS * GRU_H;
    float*    hlr = (float*)(Ye + (size_t)M_ROWS * GRU_H);
    float*    hle = hlr + GRU_B * GRU_H;

    const dim3 pgrid(H3 / 64, M_ROWS / 32);

    // Layer 0 input projections (K=130 padded to 160)
    proj_kernel<float, 160><<<pgrid, 256, 0, stream>>>(ru, GRU_I, GRU_I, ru_Wih0, ru_bih0, XPr);
    proj_kernel<float, 160><<<pgrid, 256, 0, stream>>>(en, GRU_I, GRU_I, en_Wih0, en_bih0, XPe);

    // Layer 0 scan (both streams, 32 blocks)
    gru_scan_kernel<<<32, 256, 0, stream>>>(XPr, XPe, ru_Whh0, en_Whh0,
                                            ru_bhh0, en_bhh0, Yr, Ye,
                                            nullptr, nullptr);

    // Layer 1 input projections (K=128) — XP buffers reused
    proj_kernel<_Float16, 128><<<pgrid, 256, 0, stream>>>(Yr, GRU_H, GRU_H, ru_Wih1, ru_bih1, XPr);
    proj_kernel<_Float16, 128><<<pgrid, 256, 0, stream>>>(Ye, GRU_H, GRU_H, en_Wih1, en_bih1, XPe);

    // Layer 1 scan, emit last hidden states
    gru_scan_kernel<<<32, 256, 0, stream>>>(XPr, XPe, ru_Whh1, en_Whh1,
                                            ru_bhh1, en_bhh1, nullptr, nullptr,
                                            hlr, hle);

    // Head
    mlp_kernel<<<1, 256, 0, stream>>>(hlr, hle, W1, b1, W2, b2, out);
}
